// MultiHeadSelfAttention_17291538334459
// MI455X (gfx1250) — matrix operand order repaired and checked by reading, not compiled, not run on hardware
//
#include <hip/hip_runtime.h>
#include <hip/hip_bf16.h>

// ---------------------------------------------------------------------------
// MultiHeadSelfAttention for MI455X (gfx1250, wave32, WMMA bf16)
// B=8, S=1024, D=1024, H=16, HD=64
// ---------------------------------------------------------------------------

typedef __attribute__((ext_vector_type(16))) __bf16 v16bf;
typedef __attribute__((ext_vector_type(8)))  __bf16 v8bf;
typedef __attribute__((ext_vector_type(8)))  float  v8f;

#define DEVINL __device__ __forceinline__

static __device__ const v8f kVZero = {0.f, 0.f, 0.f, 0.f, 0.f, 0.f, 0.f, 0.f};

DEVINL v8f wmma_bf16(v16bf a, v16bf b, v8f c) {
  // D = A(16x32 bf16) x B(32x16 bf16) + C(16x16 f32)
  return __builtin_amdgcn_wmma_f32_16x16x32_bf16(
      /*neg_a=*/false, a, /*neg_b=*/false, b,
      /*c_mod=*/(short)0, c, /*reuse_a=*/false, /*reuse_b=*/false);
}

// Build a 16-element A-fragment from two contiguous 8-element chunks
// (as loaded here: lane L holds row M=L&15; K = {lh*16..+7} and {lh*16+8..+15}).
DEVINL v16bf join8(v8bf lo, v8bf hi) {
  v16bf r;
#pragma unroll
  for (int i = 0; i < 8; ++i) { r[i] = lo[i]; r[i + 8] = hi[i]; }
  return r;
}

// ---------------------------------------------------------------------------
// f32 -> bf16 convert / transpose prep kernels
// ---------------------------------------------------------------------------
__global__ void __launch_bounds__(256) k_cvt_bf16(const float* __restrict__ in,
                                                  __bf16* __restrict__ out, int n) {
  int i = blockIdx.x * blockDim.x + threadIdx.x;
  if (i < n) out[i] = (__bf16)in[i];
}

// out[n*K + k] = in[k*N + n]   (in: [K,N] row-major -> out: [N,K] row-major)
__global__ void __launch_bounds__(256) k_transpose_bf16(const float* __restrict__ in,
                                                        __bf16* __restrict__ out,
                                                        int N, int K) {
  int i = blockIdx.x * blockDim.x + threadIdx.x;
  if (i < N * K) {
    int nn = i / K, kk = i % K;
    out[i] = (__bf16)in[(size_t)kk * N + nn];
  }
}

// ---------------------------------------------------------------------------
// WMMA GEMM mainloop: C_tile(128x128) = A[M,K] x Bt[N,K]^T, K-step 64,
// register-staged software pipeline (global loads for block i+1 overlap the
// 16 WMMAs of block i). 256 threads = 8 waves; wave (wm,wn) owns 64x32 =
// 4x2 fragments (A reused 2x, B reused 4x per LDS read).
// ---------------------------------------------------------------------------
DEVINL void gemm_tile(const __bf16* __restrict__ A, const __bf16* __restrict__ Bt,
                      int K, int m_base, int n_base,
                      __bf16* As, __bf16* Bs, v8f acc[4][2]) {
  const int tid  = threadIdx.x;
  const int wid  = tid >> 5;
  const int lane = tid & 31;
  const int lh   = lane >> 4;
  const int ln   = lane & 15;
  const int wm   = wid >> 2;   // 0..1 (M)
  const int wn   = wid & 3;    // 0..3 (N)

  // staging chunk coords: tile 128 rows x 64 cols, 8 bf16 per chunk,
  // 8 chunks per row, 1024 chunks, 4 per thread
  int srow[4], scol[4];
#pragma unroll
  for (int i = 0; i < 4; ++i) {
    int cc  = tid + i * 256;
    srow[i] = cc >> 3;
    scol[i] = (cc & 7) << 3;
  }

  v8bf aReg[4], bReg[4];
#pragma unroll
  for (int i = 0; i < 4; ++i) {
    aReg[i] = *(const v8bf*)&A[(size_t)(m_base + srow[i]) * K + scol[i]];
    bReg[i] = *(const v8bf*)&Bt[(size_t)(n_base + srow[i]) * K + scol[i]];
  }

  for (int k0 = 0; k0 < K; k0 += 64) {
    // land staged registers in LDS
#pragma unroll
    for (int i = 0; i < 4; ++i) {
      *(v8bf*)&As[srow[i] * 64 + scol[i]] = aReg[i];
      *(v8bf*)&Bs[srow[i] * 64 + scol[i]] = bReg[i];
    }
    __syncthreads();

    // issue global loads for the next K block; they retire at the next
    // iteration's ds_store, overlapping the WMMAs below
    if (k0 + 64 < K) {
#pragma unroll
      for (int i = 0; i < 4; ++i) {
        aReg[i] = *(const v8bf*)&A[(size_t)(m_base + srow[i]) * K + k0 + 64 + scol[i]];
        bReg[i] = *(const v8bf*)&Bt[(size_t)(n_base + srow[i]) * K + k0 + 64 + scol[i]];
      }
    }

#pragma unroll
    for (int ks = 0; ks < 2; ++ks) {
      v16bf af[4], bfr[2];
#pragma unroll
      for (int ms = 0; ms < 4; ++ms) {
        const __bf16* pa = &As[(wm * 64 + ms * 16 + ln) * 64 + ks * 32];
        af[ms] = join8(*(const v8bf*)(pa + lh * 16),
                       *(const v8bf*)(pa + lh * 16 + 8));
      }
#pragma unroll
      for (int ns = 0; ns < 2; ++ns) {
        const __bf16* pb = &Bs[(wn * 32 + ns * 16 + ln) * 64 + ks * 32 + lh * 16];
        bfr[ns] = *(const v16bf*)pb;
      }
#pragma unroll
      for (int ms = 0; ms < 4; ++ms)
#pragma unroll
        for (int ns = 0; ns < 2; ++ns)
          acc[ms][ns] = wmma_bf16(af[ms], bfr[ns], acc[ms][ns]);
    }
    __syncthreads();
  }
}

// ---------------------------------------------------------------------------
// QKV projection: xb[8192,1024] x Wqkv_t[3072,1024]^T
// Sections (q/k/v) are 1024-aligned and N tiles are 128 wide -> the section
// is block-uniform; branch once, not per element.
// Q,K -> [b,h,s,hd] bf16 ; V -> [b,h,hd,s] bf16 (transposed)
// ---------------------------------------------------------------------------
__global__ void __launch_bounds__(256) k_gemm_qkv(const __bf16* __restrict__ xb,
                                                  const __bf16* __restrict__ Wt,
                                                  __bf16* __restrict__ Qb,
                                                  __bf16* __restrict__ Kb,
                                                  __bf16* __restrict__ Vt) {
  __shared__ __align__(32) __bf16 As[128 * 64];
  __shared__ __align__(32) __bf16 Bs[128 * 64];
  v8f acc[4][2];
#pragma unroll
  for (int i = 0; i < 4; ++i)
#pragma unroll
    for (int j = 0; j < 2; ++j) acc[i][j] = kVZero;

  const int m_base = blockIdx.x * 128;
  const int n_base = blockIdx.y * 128;
  gemm_tile(xb, Wt, 1024, m_base, n_base, As, Bs, acc);

  const int lane = threadIdx.x & 31, lh = lane >> 4, ln = lane & 15;
  const int wid = threadIdx.x >> 5, wm = wid >> 2, wn = wid & 3;
  const int sec   = n_base >> 10;        // block-uniform: 0=q, 1=k, 2=v
  const int cbase = (n_base & 1023) + wn * 32;

  if (sec < 2) {
    __bf16* __restrict__ dst = (sec == 0) ? Qb : Kb;
#pragma unroll
    for (int ms = 0; ms < 4; ++ms)
#pragma unroll
      for (int ns = 0; ns < 2; ++ns) {
        const int c  = cbase + ns * 16 + ln;
        const int h  = c >> 6, hd = c & 63;
#pragma unroll
        for (int r = 0; r < 8; ++r) {
          const int m = m_base + wm * 64 + ms * 16 + r + lh * 8;
          const int b = m >> 10, s = m & 1023;
          dst[(size_t)((b * 16 + h) * 1024 + s) * 64 + hd] = (__bf16)acc[ms][ns][r];
        }
      }
  } else {
#pragma unroll
    for (int ms = 0; ms < 4; ++ms)
#pragma unroll
      for (int ns = 0; ns < 2; ++ns) {
        const int c  = cbase + ns * 16 + ln;
        const int h  = c >> 6, hd = c & 63;
#pragma unroll
        for (int r = 0; r < 8; ++r) {
          const int m = m_base + wm * 64 + ms * 16 + r + lh * 8;
          const int b = m >> 10, s = m & 1023;
          Vt[((size_t)(b * 16 + h) * 64 + hd) * 1024 + s] = (__bf16)acc[ms][ns][r];
        }
      }
  }
}

// ---------------------------------------------------------------------------
// Flash attention: 1 wave per 16-query tile, online softmax over 32-key blocks
// ---------------------------------------------------------------------------
__global__ void __launch_bounds__(128) k_attn(const __bf16* __restrict__ Qb,
                                              const __bf16* __restrict__ Kb,
                                              const __bf16* __restrict__ Vt,
                                              const float* __restrict__ mask,
                                              __bf16* __restrict__ attn) {
  __shared__ __align__(32) __bf16 Pl[4][512];   // per-wave P repack scratch
  const int tid  = threadIdx.x;
  const int w    = tid >> 5;
  const int lane = tid & 31, lh = lane >> 4, ln = lane & 15;
  const int bh   = blockIdx.x >> 4;             // b*16 + h
  const int qc   = blockIdx.x & 15;
  const int b    = bh >> 4;
  const int h    = bh & 15;
  const int q_base = qc * 64 + w * 16;
  const size_t kv_base = (size_t)bh * 1024 * 64;

  // Q A-fragments: hd 0..31 and 32..63
  v16bf qa[2];
  {
    const __bf16* qp = Qb + kv_base + (size_t)(q_base + ln) * 64;
#pragma unroll
    for (int f = 0; f < 2; ++f)
      qa[f] = join8(*(const v8bf*)(qp + f * 32 + lh * 16),
                    *(const v8bf*)(qp + f * 32 + lh * 16 + 8));
  }

  float m_i[8], l_i[8];
  v8f O[4];
#pragma unroll
  for (int r = 0; r < 8; ++r) { m_i[r] = -1e30f; l_i[r] = 0.f; }
#pragma unroll
  for (int f = 0; f < 4; ++f) O[f] = kVZero;

  const float* mrow = mask + (size_t)b * 1024;
  const float scale = 0.125f;   // 1/sqrt(64)

  for (int key0 = 0; key0 < 1024; key0 += 32) {
    // K^T B-fragments: [key subtile][hd half]
    v16bf kf[2][2];
#pragma unroll
    for (int ns = 0; ns < 2; ++ns) {
      const __bf16* kp = Kb + kv_base + (size_t)(key0 + ns * 16 + ln) * 64;
      kf[ns][0] = *(const v16bf*)(kp + lh * 16);
      kf[ns][1] = *(const v16bf*)(kp + 32 + lh * 16);
    }
    v8f s0 = kVZero, s1 = kVZero;
    s0 = wmma_bf16(qa[0], kf[0][0], s0);
    s0 = wmma_bf16(qa[1], kf[0][1], s0);
    s1 = wmma_bf16(qa[0], kf[1][0], s1);
    s1 = wmma_bf16(qa[1], kf[1][1], s1);

    const float mk0 = mrow[key0 + ln];
    const float mk1 = mrow[key0 + 16 + ln];
    const float bias0 = -(1.0f - mk0) * 1e10f;
    const float bias1 = -(1.0f - mk1) * 1e10f;

    float bm[8];
#pragma unroll
    for (int r = 0; r < 8; ++r) {
      float a0 = s0[r] * scale + bias0;
      float a1 = s1[r] * scale + bias1;
      s0[r] = a0; s1[r] = a1;
      bm[r] = fmaxf(a0, a1);
    }
    // row-max across the 16 lanes holding this row's columns
#pragma unroll
    for (int off = 1; off < 16; off <<= 1)
#pragma unroll
      for (int r = 0; r < 8; ++r)
        bm[r] = fmaxf(bm[r], __shfl_xor(bm[r], off, 32));

    float alpha[8], rs[8];
#pragma unroll
    for (int r = 0; r < 8; ++r) {
      float mn = fmaxf(m_i[r], bm[r]);
      alpha[r] = __expf(m_i[r] - mn);
      m_i[r] = mn;
      float p0 = __expf(s0[r] - mn) * mk0;   // probas *= mask
      float p1 = __expf(s1[r] - mn) * mk1;
      s0[r] = p0; s1[r] = p1;
      rs[r] = p0 + p1;
    }
#pragma unroll
    for (int off = 1; off < 16; off <<= 1)
#pragma unroll
      for (int r = 0; r < 8; ++r)
        rs[r] += __shfl_xor(rs[r], off, 32);
#pragma unroll
    for (int r = 0; r < 8; ++r)
      l_i[r] = l_i[r] * alpha[r] + rs[r];
#pragma unroll
    for (int f = 0; f < 4; ++f)
#pragma unroll
      for (int r = 0; r < 8; ++r)
        O[f][r] *= alpha[r];

    // Repack P (C layout, f32) -> A layout (bf16) via per-wave LDS slice.
    // DS ops from one wave execute in order, so no barrier is needed.
    __bf16* pl = &Pl[w][0];
#pragma unroll
    for (int r = 0; r < 8; ++r) {
      const int row = r + lh * 8;
      pl[row * 32 + ln]      = (__bf16)s0[r];
      pl[row * 32 + 16 + ln] = (__bf16)s1[r];
    }
    v16bf pa;
    {
      const __bf16* pp = pl + ln * 32;
      pa = join8(*(const v8bf*)(pp + lh * 16), *(const v8bf*)(pp + lh * 16 + 8));
    }

    // O += P(16x32) x V(32x64); V stored [hd, s] so B-fragments are contiguous
#pragma unroll
    for (int f = 0; f < 4; ++f) {
      const __bf16* vp = Vt + ((size_t)bh * 64 + f * 16 + ln) * 1024 + key0 + lh * 16;
      v16bf vb = *(const v16bf*)vp;
      O[f] = wmma_bf16(pa, vb, O[f]);
    }
    if (key0 + 32 < 1024)
      __builtin_prefetch(Kb + kv_base + (size_t)(key0 + 32 + ln) * 64, 0, 1);
  }

  float inv[8];
#pragma unroll
  for (int r = 0; r < 8; ++r) inv[r] = 1.0f / fmaxf(l_i[r], 1e-20f);
#pragma unroll
  for (int f = 0; f < 4; ++f)
#pragma unroll
    for (int r = 0; r < 8; ++r) {
      const int s = q_base + r + lh * 8;
      attn[(size_t)(b * 1024 + s) * 1024 + h * 64 + f * 16 + ln] =
          (__bf16)(O[f][r] * inv[r]);
    }
}

// ---------------------------------------------------------------------------
// Output projection: attn[8192,1024] x Wproj_t[1024,1024]^T + bias -> f32
// ---------------------------------------------------------------------------
__global__ void __launch_bounds__(256) k_gemm_proj(const __bf16* __restrict__ Ab,
                                                   const __bf16* __restrict__ Wt,
                                                   const float* __restrict__ bias,
                                                   float* __restrict__ out) {
  __shared__ __align__(32) __bf16 As[128 * 64];
  __shared__ __align__(32) __bf16 Bs[128 * 64];
  v8f acc[4][2];
#pragma unroll
  for (int i = 0; i < 4; ++i)
#pragma unroll
    for (int j = 0; j < 2; ++j) acc[i][j] = kVZero;

  const int m_base = blockIdx.x * 128;
  const int n_base = blockIdx.y * 128;
  gemm_tile(Ab, Wt, 1024, m_base, n_base, As, Bs, acc);

  const int lane = threadIdx.x & 31, lh = lane >> 4, ln = lane & 15;
  const int wid = threadIdx.x >> 5, wm = wid >> 2, wn = wid & 3;
#pragma unroll
  for (int ms = 0; ms < 4; ++ms)
#pragma unroll
    for (int ns = 0; ns < 2; ++ns) {
      const int n = n_base + wn * 32 + ns * 16 + ln;
      const float bn = bias[n];
#pragma unroll
      for (int r = 0; r < 8; ++r) {
        const int m = m_base + wm * 64 + ms * 16 + r + lh * 8;
        out[(size_t)m * 1024 + n] = acc[ms][ns][r] + bn;
      }
    }
}

// ---------------------------------------------------------------------------
extern "C" void kernel_launch(void* const* d_in, const int* in_sizes, int n_in,
                              void* d_out, int out_size, void* d_ws, size_t ws_size,
                              hipStream_t stream) {
  (void)in_sizes; (void)n_in; (void)out_size; (void)ws_size;
  const float* x     = (const float*)d_in[0];
  const float* pmask = (const float*)d_in[1];   // [B,1,S] -> flat B*S
  const float* Wqkv  = (const float*)d_in[2];   // [1024, 3072]
  const float* Wproj = (const float*)d_in[3];   // [1024, 1024]
  const float* bproj = (const float*)d_in[4];   // [1024]
  float* out = (float*)d_out;                   // [8,1024,1024] f32

  char* ws = (char*)d_ws;
  size_t off = 0;
  auto alloc = [&](size_t bytes) -> void* {
    void* p = ws + off;
    off = (off + bytes + 255) & ~(size_t)255;
    return p;
  };
  __bf16* xb   = (__bf16*)alloc(8192ull * 1024 * 2);        // 16 MB
  __bf16* Wqt  = (__bf16*)alloc(3072ull * 1024 * 2);        //  6 MB
  __bf16* Wpt  = (__bf16*)alloc(1024ull * 1024 * 2);        //  2 MB
  __bf16* Qb   = (__bf16*)alloc(8ull * 16 * 1024 * 64 * 2); // 16 MB
  __bf16* Kb   = (__bf16*)alloc(8ull * 16 * 1024 * 64 * 2); // 16 MB
  __bf16* Vt   = (__bf16*)alloc(8ull * 16 * 64 * 1024 * 2); // 16 MB
  __bf16* attn = (__bf16*)alloc(8192ull * 1024 * 2);        // 16 MB

  // prep: bf16 convert + weight transposes
  k_cvt_bf16<<<32768, 256, 0, stream>>>(x, xb, 8192 * 1024);
  k_transpose_bf16<<<12288, 256, 0, stream>>>(Wqkv, Wqt, 3072, 1024);
  k_transpose_bf16<<<4096, 256, 0, stream>>>(Wproj, Wpt, 1024, 1024);

  // QKV projection (M: 64 tiles of 128, N: 24 tiles of 128)
  k_gemm_qkv<<<dim3(64, 24), 256, 0, stream>>>(xb, Wqt, Qb, Kb, Vt);

  // flash attention: B*H * (S/64) blocks, 4 waves each
  k_attn<<<8 * 16 * 16, 128, 0, stream>>>(Qb, Kb, Vt, pmask, attn);

  // output projection (M: 64 tiles of 128, N: 8 tiles of 128)
  k_gemm_proj<<<dim3(64, 8), 256, 0, stream>>>(attn, Wpt, bproj, out);
}
